// SharedRoutedMoE_BHWC_16939351015742
// MI455X (gfx1250) — matrix operand order repaired and checked by reading, not compiled, not run on hardware
//
#include <hip/hip_runtime.h>
#include <hip/hip_bf16.h>

typedef __bf16 v16bf __attribute__((ext_vector_type(16)));
typedef float  v8f   __attribute__((ext_vector_type(8)));
typedef unsigned int u32x4 __attribute__((ext_vector_type(4)));
typedef int          i32x8 __attribute__((ext_vector_type(8)));
typedef int          i32x4 __attribute__((ext_vector_type(4)));

union V16U { v16bf v; uint4 q[2]; };
union V8FU { v8f v; float f[8]; };

#define TC   512     // channels
#define THD  1024    // hidden
#define TBM  64      // tokens per block
#define LDX  520     // lds_x row stride (bf16): 512 data + 8 pad (16B) -> matches TDM pad
#define LDH  1032    // lds_h row stride (bf16 elems), 2064B

#if defined(__has_builtin)
#if __has_builtin(__builtin_amdgcn_tensor_load_to_lds)
#define HAVE_TDM 1
#endif
#endif

// ---------------------------------------------------------------------------
// Weight transpose + fp32 -> bf16 convert.  src is [R][Cc] row-major fp32,
// dst is [Cc][R] row-major bf16.
// ---------------------------------------------------------------------------
__global__ void k_transpose_bf16(const float* __restrict__ src,
                                 __bf16* __restrict__ dst, int R, int Cc) {
  int idx = blockIdx.x * blockDim.x + threadIdx.x;
  if (idx >= R * Cc) return;
  int r = idx / Cc;
  int c = idx - r * Cc;
  dst[(size_t)c * R + r] = (__bf16)src[idx];
}

// ---------------------------------------------------------------------------
// Router: one wave32 per token.  Fused: x fp32 -> bf16 convert, logits,
// softmax (E=2), argmax.  Writes gates (wg) and one_hot*gate (wsel).
// ---------------------------------------------------------------------------
__global__ void k_router(const float* __restrict__ x, const float* __restrict__ rw,
                         const float* __restrict__ rb, __bf16* __restrict__ xb,
                         float* __restrict__ wsel, float* __restrict__ wg, int N) {
  const int wave = threadIdx.x >> 5;
  const int lane = threadIdx.x & 31;
  const int n = blockIdx.x * (blockDim.x >> 5) + wave;
  if (n >= N) return;

  const float4* xp = (const float4*)(x + (size_t)n * TC + lane * 16);
  const float4* rp = (const float4*)(rw + lane * 32);
  float a0 = 0.f, a1 = 0.f;
  __attribute__((aligned(16))) __bf16 tmp[16];
#pragma unroll
  for (int q = 0; q < 4; ++q) {
    float4 xv = xp[q];
    float4 r0 = rp[2 * q];
    float4 r1 = rp[2 * q + 1];
    a0 += xv.x * r0.x + xv.y * r0.z + xv.z * r1.x + xv.w * r1.z;
    a1 += xv.x * r0.y + xv.y * r0.w + xv.z * r1.y + xv.w * r1.w;
    tmp[4 * q + 0] = (__bf16)xv.x;
    tmp[4 * q + 1] = (__bf16)xv.y;
    tmp[4 * q + 2] = (__bf16)xv.z;
    tmp[4 * q + 3] = (__bf16)xv.w;
  }
  uint4* xo = (uint4*)(xb + (size_t)n * TC + lane * 16);
  xo[0] = *(const uint4*)&tmp[0];
  xo[1] = *(const uint4*)&tmp[8];

#pragma unroll
  for (int off = 16; off > 0; off >>= 1) {
    a0 += __shfl_xor(a0, off, 32);
    a1 += __shfl_xor(a1, off, 32);
  }
  if (lane == 0) {
    float l0 = a0 + rb[0], l1 = a1 + rb[1];
    float m = fmaxf(l0, l1);
    float e0 = __expf(l0 - m), e1 = __expf(l1 - m);
    float s = e0 + e1;
    float g0 = e0 / s, g1 = e1 / s;
    int arg = (g0 >= g1) ? 0 : 1;  // ties -> first, matches jnp.argmax
    wg[2 * n] = g0;
    wg[2 * n + 1] = g1;
    wsel[2 * n] = (arg == 0) ? g0 : 0.f;
    wsel[2 * n + 1] = (arg == 1) ? g1 : 0.f;
  }
}

// ---------------------------------------------------------------------------
// Fused MoE: per 64-token tile run 3 MLPs (shared, expert0, expert1).
// X tile staged into LDS by the Tensor Data Mover (row-padded layout done in
// HW via D# pad_interval/pad_amount).  GEMM1 -> GELU -> gate-scale -> bf16 H
// in LDS; GEMM2 accumulates into one persistent fp32 accumulator.
// ---------------------------------------------------------------------------
__global__ void __launch_bounds__(256)
k_moe_main(const __bf16* __restrict__ xb, const __bf16* __restrict__ wT,
           const float* __restrict__ sb1, const float* __restrict__ eb1,
           const float* __restrict__ sb2, const float* __restrict__ eb2,
           const float* __restrict__ wsel, float* __restrict__ out) {
  extern __shared__ char smem[];
  __bf16* lds_x = (__bf16*)smem;                               // TBM*LDX (at dyn-LDS base)
  __bf16* lds_h = (__bf16*)(smem + (size_t)TBM * LDX * 2);     // TBM*LDH
  float* sel = (float*)(smem + (size_t)TBM * LDX * 2 + (size_t)TBM * LDH * 2);

  const int tid = threadIdx.x;
  const int wave = tid >> 5;
  const int lane = tid & 31;
  const int l15 = lane & 15;
  const int hf = lane >> 4;  // lane-half selector for WMMA operand layout
  const size_t m0 = (size_t)blockIdx.x * TBM;

#ifdef HAVE_TDM
  // ---- Stage X tile (64 x 512 bf16) via Tensor Data Mover ----
  // D# describes a 2D bf16 tensor tile 512x64 with LDS row padding of
  // 4 DWORDs (16B) after every 256 DWORDs (one 1024B row) -> LDX=520 layout.
  if (wave == 0) {
    unsigned ldsbase = __builtin_amdgcn_groupstaticsize();  // dyn LDS starts here
    unsigned long long ga = (unsigned long long)(const void*)(xb + m0 * TC);
    u32x4 g0 = {};
    g0.x = 1u;                                              // count=1, user D#
    g0.y = ldsbase;                                         // lds_addr
    g0.z = (unsigned)ga;                                    // global_addr[31:0]
    g0.w = (unsigned)((ga >> 32) & 0x01FFFFFFu) | (2u << 30);  // addr[56:32], type=2
    i32x8 g1 = {};
    g1[0] = (int)((1u << 16)      // data_size = 2 bytes
                  | (1u << 20)    // pad_enable
                  | (7u << 22)    // pad_interval: 256 DWORDs (= one row)
                  | (3u << 25));  // pad_amount: 4 DWORDs (16B)
    g1[1] = (int)(512u << 16);    // tensor_dim0 = 512
    g1[2] = (int)(64u << 16);     // tensor_dim1 = 64
    g1[3] = (int)(512u << 16);    // tile_dim0 = 512
    g1[4] = 64;                   // tile_dim1 = 64 (tile_dim2 = 0)
    g1[5] = 512;                  // tensor_dim0_stride = 512
    g1[6] = 0;
    g1[7] = 0;
    i32x4 gz4 = {};
    i32x8 gz8 = {};
    __builtin_amdgcn_tensor_load_to_lds(g0, g1, gz4, gz4, gz8, 0);
    __builtin_amdgcn_s_wait_tensorcnt(0);
  }
#else
  for (int i = tid; i < TBM * TC / 8; i += 256) {
    int row = i >> 6;
    int q = i & 63;
    *(uint4*)(lds_x + row * LDX + q * 8) =
        *(const uint4*)(xb + (m0 + row) * TC + q * 8);
  }
#endif
  if (tid < TBM * 2) sel[tid] = wsel[m0 * 2 + tid];
  __syncthreads();

  const size_t HdC = (size_t)THD * TC;
  v8f vzero = {};
  v8f acc2[16];  // 4 row tiles x 4 col tiles of the 64x512 output, per wave
#pragma unroll
  for (int t = 0; t < 16; ++t) acc2[t] = vzero;

  for (int mlp = 0; mlp < 3; ++mlp) {
    const __bf16* w1T = wT + (size_t)mlp * HdC;        // [THD][TC] (= W1^T)
    const __bf16* w2T = wT + (size_t)(3 + mlp) * HdC;  // [TC][THD] (= W2^T)
    const float* b1 = (mlp == 0) ? sb1 : (eb1 + (size_t)(mlp - 1) * THD);

    // ---- GEMM1: H[64x1024]; wave owns 8 col-tiles, A fragment hoisted ----
    for (int i = 0; i < 4; ++i) {
      V8FU acc1[8];
#pragma unroll
      for (int jj = 0; jj < 8; ++jj) acc1[jj].v = vzero;
      const __bf16* arow = lds_x + (i * 16 + l15) * LDX + hf * 8;
      for (int k = 0; k < TC; k += 32) {
        V16U a;
        a.q[0] = *(const uint4*)(arow + k);       // K 0-7 / 8-15
        a.q[1] = *(const uint4*)(arow + k + 16);  // K 16-23 / 24-31
#pragma unroll
        for (int jj = 0; jj < 8; ++jj) {
          const int n0 = (wave * 8 + jj) * 16;
          V16U b;
          const __bf16* brow = w1T + (size_t)(n0 + l15) * TC + hf * 8 + k;
          b.q[0] = *(const uint4*)(brow);
          b.q[1] = *(const uint4*)(brow + 16);
          acc1[jj].v = __builtin_amdgcn_wmma_f32_16x16x32_bf16(
              false, a.v, false, b.v, (short)0, acc1[jj].v, false, false);
        }
      }
      // epilogue: bias + exact GELU + gate scale -> bf16 H in LDS
#pragma unroll
      for (int jj = 0; jj < 8; ++jj) {
        const int col = (wave * 8 + jj) * 16 + l15;
        const float b1v = b1[col];
#pragma unroll
        for (int r = 0; r < 8; ++r) {
          int lrow = i * 16 + hf * 8 + r;
          float vv = acc1[jj].f[r] + b1v;
          float g = 0.5f * vv * (1.0f + erff(vv * 0.70710678118f));
          float sc = (mlp == 0) ? 1.0f : sel[lrow * 2 + (mlp - 1)];
          lds_h[lrow * LDH + col] = (__bf16)(g * sc);
        }
      }
    }
    __syncthreads();

    // ---- GEMM2: acc2 += H @ W2 ; wave owns 4 col-tiles x 4 row-tiles ----
    for (int k = 0; k < THD; k += 32) {
#pragma unroll
      for (int i = 0; i < 4; ++i) {
        V16U a;
        const __bf16* arow = lds_h + (i * 16 + l15) * LDH + hf * 8 + k;
        a.q[0] = *(const uint4*)(arow);
        a.q[1] = *(const uint4*)(arow + 16);
#pragma unroll
        for (int jj = 0; jj < 4; ++jj) {
          const int n0 = (wave * 4 + jj) * 16;
          V16U b;
          const __bf16* brow = w2T + (size_t)(n0 + l15) * THD + k + hf * 8;
          b.q[0] = *(const uint4*)(brow);
          b.q[1] = *(const uint4*)(brow + 16);
          acc2[i * 4 + jj] = __builtin_amdgcn_wmma_f32_16x16x32_bf16(
              false, a.v, false, b.v, (short)0, acc2[i * 4 + jj], false, false);
        }
      }
    }
    __syncthreads();  // lds_h reused by next MLP
  }

  // ---- Epilogue: + sb2 + sel0*eb2[0] + sel1*eb2[1], store fp32 ----
#pragma unroll
  for (int i = 0; i < 4; ++i) {
#pragma unroll
    for (int jj = 0; jj < 4; ++jj) {
      const int col = (wave * 4 + jj) * 16 + l15;
      const float bs = sb2[col];
      const float be0 = eb2[col];
      const float be1 = eb2[TC + col];
      V8FU a;
      a.v = acc2[i * 4 + jj];
#pragma unroll
      for (int r = 0; r < 8; ++r) {
        int lrow = i * 16 + hf * 8 + r;
        float o = a.f[r] + bs + sel[lrow * 2] * be0 + sel[lrow * 2 + 1] * be1;
        out[(m0 + lrow) * TC + col] = o;
      }
    }
  }
}

// ---------------------------------------------------------------------------
// Balance loss: deterministic single-block reduction (no float atomics).
// ---------------------------------------------------------------------------
__global__ void k_loss(const float* __restrict__ wg, const float* __restrict__ wsel,
                       float* __restrict__ out, int N) {
  __shared__ float s0[256], s1[256], s2[256], s3[256];
  const int tid = threadIdx.x;
  float i0 = 0.f, i1 = 0.f, d0 = 0.f, d1 = 0.f;
  for (int n = tid; n < N; n += 256) {
    i0 += wg[2 * n];
    i1 += wg[2 * n + 1];
    d0 += (wsel[2 * n] > 0.f) ? 1.f : 0.f;
    d1 += (wsel[2 * n + 1] > 0.f) ? 1.f : 0.f;
  }
  s0[tid] = i0; s1[tid] = i1; s2[tid] = d0; s3[tid] = d1;
  __syncthreads();
  for (int off = 128; off > 0; off >>= 1) {
    if (tid < off) {
      s0[tid] += s0[tid + off];
      s1[tid] += s1[tid + off];
      s2[tid] += s2[tid + off];
      s3[tid] += s3[tid + off];
    }
    __syncthreads();
  }
  if (tid == 0) {
    float invN = 1.0f / (float)N;
    out[(size_t)N * TC] =
        2.0f * ((s0[0] * invN) * (s2[0] * invN) + (s1[0] * invN) * (s3[0] * invN));
  }
}

// ---------------------------------------------------------------------------
extern "C" void kernel_launch(void* const* d_in, const int* in_sizes, int n_in,
                              void* d_out, int out_size, void* d_ws, size_t ws_size,
                              hipStream_t stream) {
  const float* x   = (const float*)d_in[0];
  const float* sw1 = (const float*)d_in[1];
  const float* sb1 = (const float*)d_in[2];
  const float* sw2 = (const float*)d_in[3];
  const float* sb2 = (const float*)d_in[4];
  const float* ew1 = (const float*)d_in[5];
  const float* eb1 = (const float*)d_in[6];
  const float* ew2 = (const float*)d_in[7];
  const float* eb2 = (const float*)d_in[8];
  const float* rw  = (const float*)d_in[9];
  const float* rb  = (const float*)d_in[10];
  float* out = (float*)d_out;

  const int N = 8 * 64 * 64;  // 32768 tokens

  // workspace carve-up
  char* ws = (char*)d_ws;
  __bf16* xb = (__bf16*)ws;           ws += (size_t)N * TC * 2;   // 32 MB
  float* wsel = (float*)ws;           ws += (size_t)N * 2 * 4;    // 256 KB
  float* wg = (float*)ws;             ws += (size_t)N * 2 * 4;    // 256 KB
  __bf16* wT = (__bf16*)ws;           // 6 * 512*1024 bf16 = 6 MB

  const size_t HdC = (size_t)THD * TC;
  const int nel = TC * THD;  // 524288 elems per weight matrix

  // W1^T blocks [THD][TC] (src [TC][THD]), W2^T blocks [TC][THD] (src [THD][TC])
  k_transpose_bf16<<<nel / 256, 256, 0, stream>>>(sw1,            wT + 0 * HdC, TC, THD);
  k_transpose_bf16<<<nel / 256, 256, 0, stream>>>(ew1 + 0 * nel,  wT + 1 * HdC, TC, THD);
  k_transpose_bf16<<<nel / 256, 256, 0, stream>>>(ew1 + 1 * nel,  wT + 2 * HdC, TC, THD);
  k_transpose_bf16<<<nel / 256, 256, 0, stream>>>(sw2,            wT + 3 * HdC, THD, TC);
  k_transpose_bf16<<<nel / 256, 256, 0, stream>>>(ew2 + 0 * nel,  wT + 4 * HdC, THD, TC);
  k_transpose_bf16<<<nel / 256, 256, 0, stream>>>(ew2 + 1 * nel,  wT + 5 * HdC, THD, TC);

  // router + bf16 conversion of x
  k_router<<<N / 8, 256, 0, stream>>>(x, rw, rb, xb, wsel, wg, N);

  // fused MoE main (dynamic LDS: x tile + H tile + sel)
  const size_t smem = (size_t)TBM * LDX * 2 + (size_t)TBM * LDH * 2 + TBM * 2 * 4;
  k_moe_main<<<N / TBM, 256, smem, stream>>>(xb, wT, sb1, eb1, sb2, eb2, wsel, out);

  // balance loss scalar
  k_loss<<<1, 256, 0, stream>>>(wg, wsel, out, N);
}
